// SeqTagger_24103356465442
// MI455X (gfx1250) — matrix operand order repaired and checked by reading, not compiled, not run on hardware
//
#include <hip/hip_runtime.h>
#include <stdint.h>
#include <stddef.h>

// ---------------- problem dims ----------------
#define Bb   128
#define Tt   256
#define Vv   50000
#define Ee   300
#define Epad 320
#define Hh   512
#define Gg   2048
#define Cc   11
#define Cpad 16
#define BT   (Bb*Tt)
#define XSTR 1024          // row stride (halves) of activation buffers
#define START_TAG 9
#define STOP_TAG  10
#define NEGVAL   -100000.0f

typedef _Float16 htype;
typedef __attribute__((ext_vector_type(16))) _Float16 v16h;
typedef __attribute__((ext_vector_type(8)))  _Float16 v8h;
typedef __attribute__((ext_vector_type(8)))  float    v8f;
typedef __attribute__((ext_vector_type(4)))  unsigned int v4u;
typedef __attribute__((ext_vector_type(8)))  int v8i_t;
typedef __attribute__((ext_vector_type(4)))  int v4i_t;

// ---------------- TDM availability (device pass only) ----------------
#if defined(__AMDGCN__) && defined(__has_builtin)
#if __has_builtin(__builtin_amdgcn_tensor_load_to_lds) && __has_builtin(__builtin_amdgcn_s_wait_tensorcnt)
#define HAS_TDM 1
#endif
#endif
#ifndef HAS_TDM
#define HAS_TDM 0
#endif

#if HAS_TDM
// Build a 2D tensor-DMA descriptor (D#) per CDNA5 ISA and
// issue TENSOR_LOAD_TO_LDS. Copies tileH rows x tileW elements (2B each) into
// LDS at ldsOff, rows packed with pitch tileW. Columns >= validCols read as 0
// (TDM OOB-zero), which gives us free zero-padding for K=300 weight rows.
// This toolchain exposes the 6-arg builtin form:
//   (uint32x4 g0, int32x8 g1, int32x4 g2, int32x4 g3, int32x8 extra, i32 cpol)
__device__ inline void tdm_load_2d(const htype* gsrc, unsigned ldsOff,
                                   int validCols, int rows, int rowStrideElems,
                                   int tileW, int tileH) {
  unsigned long long ga = (unsigned long long)(uintptr_t)gsrc;
  unsigned td0 = (unsigned)validCols;              // tensor_dim0 (elements)
  unsigned td1 = (unsigned)rows;                   // tensor_dim1 (rows)
  unsigned long long s0 = (unsigned long long)rowStrideElems; // dim0 stride
  v4u g0 = { 1u,                                   // count=1, user descriptor
             ldsOff,                               // lds_addr [63:32]
             (unsigned)(ga & 0xFFFFFFFFull),       // global_addr lo
             (unsigned)((ga >> 32) & 0x01FFFFFFull) | (2u << 30) }; // hi + type=2
  v8i_t g1 = { (int)(1u << 16),                                         // data_size=2B
               (int)((td0 & 0xFFFFu) << 16),                            // dim0 lo16
               (int)(((td0 >> 16) & 0xFFFFu) | ((td1 & 0xFFFFu) << 16)),// dim0 hi / dim1 lo
               (int)(((td1 >> 16) & 0xFFFFu) | (((unsigned)tileW & 0xFFFFu) << 16)),
               (int)((unsigned)tileH & 0xFFFFu),                        // tile_dim1, tile_dim2=0
               (int)(unsigned)(s0 & 0xFFFFFFFFull),                     // dim0_stride lo32
               (int)(unsigned)((s0 >> 32) & 0xFFFFu),                   // stride hi / dim1_stride=0
               0 };
  v4i_t g2 = {0,0,0,0};
  v4i_t g3 = {0,0,0,0};
  v8i_t g4 = {0,0,0,0,0,0,0,0};
  __builtin_amdgcn_tensor_load_to_lds(g0, g1, g2, g3, g4, 0);
}
#endif

// ---------------- WMMA fragment helpers (contiguous k order, same for A and B) ----------------
// A (16x32 f16): lane<16 row=m0+lane holds K k0..k0+15;
//                lane>=16 same row group, K k0+16..k0+31.
__device__ inline v16h frag_a(const htype* base, int lda, int m0, int k0, int lane) {
  int row = m0 + (lane & 15);
  int kb  = k0 + ((lane >> 4) << 4);
  const htype* p = base + (size_t)row * lda + kb;
  v8h lo = *(const v8h*)p;
  v8h hi = *(const v8h*)(p + 8);
  return __builtin_shufflevector(lo, hi, 0,1,2,3,4,5,6,7,8,9,10,11,12,13,14,15);
}
// B (32x16 f16) from row-major weights W[N][K] (column n of B == row n of W):
// lane<16 col=n0+lane holds K k0..k0+15; lane>=16 holds K k0+16..k0+31.
__device__ inline v16h frag_b_g(const htype* base, int ldb, int n0, int k0, int lane) {
  int row = n0 + (lane & 15);
  int kb  = k0 + ((lane >> 4) << 4);
  const htype* p = base + (size_t)row * ldb + kb;
  v8h lo = *(const v8h*)p;
  v8h hi = *(const v8h*)(p + 8);
  return __builtin_shufflevector(lo, hi, 0,1,2,3,4,5,6,7,8,9,10,11,12,13,14,15);
}

// ---------------- input-projection GEMM: GX = X(M x K) * W^T + bias ----------------
// Block = 8 waves arranged 2(M) x 4(N): 32x64 output tile. Weight slab staged
// to LDS by the Tensor Data Mover; B fragments come from LDS (ds_load),
// A fragments straight from global (X stays L2-resident).
template <int KB, int KPAD>
__global__ void __launch_bounds__(256)
gemm_gx(const htype* __restrict__ X, const htype* __restrict__ W,
        const float* __restrict__ bias, htype* __restrict__ GX, int Kw) {
  __shared__ __align__(16) htype smem[64 * KB];
  const int tid = threadIdx.x, wave = tid >> 5, lane = tid & 31;
  const int m0   = blockIdx.x * 32 + (wave >> 2) * 16;
  const int nblk = blockIdx.y * 64;
  const int nloc = (wave & 3) * 16;
  const int col  = nblk + nloc + (lane & 15);
  float bv = bias[col];
  v8f acc;
#pragma unroll
  for (int r = 0; r < 8; ++r) acc[r] = bv;

  for (int kb = 0; kb < KPAD; kb += KB) {
    __syncthreads();
#if HAS_TDM
    if (wave == 0) {
      unsigned ldsOff = (unsigned)(uintptr_t)(&smem[0]);
      tdm_load_2d(W + (size_t)nblk * Kw + kb, ldsOff, Kw - kb, 64, Kw, KB, 64);
      __builtin_amdgcn_s_wait_tensorcnt(0);
    }
#else
    for (int i = tid; i < 64 * KB; i += 256) {
      int r = i / KB, c = i % KB, gc = kb + c;
      smem[i] = (gc < Kw) ? W[(size_t)(nblk + r) * Kw + gc] : (htype)0.f;
    }
#endif
    __syncthreads();
    for (int kk = 0; kk < KB; kk += 32) {
      v16h a = frag_a(X, XSTR, m0, kb + kk, lane);
      int brow = nloc + (lane & 15);
      int bk   = kk + ((lane >> 4) << 4);
      const htype* bp = &smem[brow * KB + bk];
      v8h blo = *(const v8h*)bp;
      v8h bhi = *(const v8h*)(bp + 8);
      v16h b = __builtin_shufflevector(blo, bhi, 0,1,2,3,4,5,6,7,8,9,10,11,12,13,14,15);
      acc = __builtin_amdgcn_wmma_f32_16x16x32_f16(false, a, false, b, (short)0, acc,
                                                   false, false);
    }
  }
  // C/D layout: lane<16 col, rows m0+0..7; lane>=16 col, rows m0+8..15
  int rbase = m0 + ((lane >> 4) << 3);
#pragma unroll
  for (int r = 0; r < 8; ++r)
    GX[(size_t)(rbase + r) * Gg + col] = (htype)acc[r];
}

// ---------------- per-step recurrent GEMM: gates = gx[:,t] + h * Whh^T ----------------
__global__ void __launch_bounds__(256)
gemm_rec(const htype* __restrict__ h16, const htype* __restrict__ Whh,
         const htype* __restrict__ GX, float* __restrict__ gates, int t_eff) {
  const int tid = threadIdx.x, wave = tid >> 5, lane = tid & 31;
  const int m0 = blockIdx.x * 16;
  const int n0 = blockIdx.y * 128 + wave * 16;
  const int col = n0 + (lane & 15);
  const int rbase = m0 + ((lane >> 4) << 3);
  v8f acc;
#pragma unroll
  for (int r = 0; r < 8; ++r)
    acc[r] = (float)GX[((size_t)(rbase + r) * Tt + t_eff) * Gg + col];
  for (int k = 0; k < Hh; k += 32) {
    v16h a = frag_a(h16, Hh, m0, k, lane);
    v16h b = frag_b_g(Whh, Hh, n0, k, lane);
    acc = __builtin_amdgcn_wmma_f32_16x16x32_f16(false, a, false, b, (short)0, acc,
                                                 false, false);
  }
#pragma unroll
  for (int r = 0; r < 8; ++r)
    gates[(size_t)(rbase + r) * Gg + col] = acc[r];
}

// ---------------- LSTM cell elementwise (one step, one direction) ----------------
__global__ void lstm_cell(const float* __restrict__ gates, float* __restrict__ hs,
                          float* __restrict__ cs, htype* __restrict__ h16,
                          htype* __restrict__ outX, const int* __restrict__ seq_len,
                          int t_eff, int dir) {
  int idx = blockIdx.x * blockDim.x + threadIdx.x;   // 0 .. B*H-1
  int b = idx >> 9, j = idx & 511;
  const float* g = gates + (size_t)b * Gg;
  float i_ = 1.f / (1.f + expf(-g[j]));
  float f_ = 1.f / (1.f + expf(-g[512 + j]));
  float gt = tanhf(g[1024 + j]);
  float o_ = 1.f / (1.f + expf(-g[1536 + j]));
  float c_old = cs[idx];
  float c_new = f_ * c_old + i_ * gt;
  float h_new = o_ * tanhf(c_new);
  bool  mt = t_eff < seq_len[b];
  float h = mt ? h_new : hs[idx];
  float c = mt ? c_new : c_old;
  hs[idx] = h; cs[idx] = c; h16[idx] = (htype)h;
  outX[(size_t)(b * Tt + t_eff) * XSTR + dir * Hh + j] = mt ? (htype)h_new : (htype)0.f;
}

// ---------------- FC emissions GEMM: emis = X * fcw^T + fcb (N padded to 16) ----------------
__global__ void __launch_bounds__(256)
gemm_fc(const htype* __restrict__ X, const htype* __restrict__ Wfc,
        const float* __restrict__ b16, float* __restrict__ emis) {
  const int tid = threadIdx.x, wave = tid >> 5, lane = tid & 31;
  const int m0 = blockIdx.x * 128 + wave * 16;
  const int col = lane & 15;
  const int rbase = m0 + ((lane >> 4) << 3);
  float bv = b16[col];
  v8f acc;
#pragma unroll
  for (int r = 0; r < 8; ++r) acc[r] = bv;
  for (int k = 0; k < 1024; k += 32) {
    v16h a = frag_a(X, XSTR, m0, k, lane);
    v16h b = frag_b_g(Wfc, 1024, 0, k, lane);
    acc = __builtin_amdgcn_wmma_f32_16x16x32_f16(false, a, false, b, (short)0, acc,
                                                 false, false);
  }
#pragma unroll
  for (int r = 0; r < 8; ++r)
    emis[(size_t)(rbase + r) * Cpad + col] = acc[r];
}

// ---------------- small prep kernels ----------------
__global__ void conv_f16(const float* __restrict__ s, htype* __restrict__ d, int n) {
  for (int i = blockIdx.x * blockDim.x + threadIdx.x; i < n; i += gridDim.x * blockDim.x)
    d[i] = (htype)s[i];
}
__global__ void bias_sum(const float* a, const float* b, float* d) {
  int i = blockIdx.x * blockDim.x + threadIdx.x;
  if (i < Gg) d[i] = a[i] + b[i];
}
__global__ void pad_fcw(const float* w, htype* d) {
  int i = blockIdx.x * blockDim.x + threadIdx.x;   // 16*1024 threads
  int r = i >> 10, k = i & 1023;
  d[i] = (r < Cc) ? (htype)w[r * 1024 + k] : (htype)0.f;
}
__global__ void pad_fcb(const float* b, float* d) {
  int j = threadIdx.x;
  if (j < Cpad) d[j] = (j < Cc) ? b[j] : 0.f;
}
__global__ void gather_emb(const int* __restrict__ tokens, const float* __restrict__ emb,
                           htype* __restrict__ X) {
  int bt = blockIdx.x, e = threadIdx.x;            // block = 320 threads
  int tok = tokens[bt];
  X[(size_t)bt * XSTR + e] = (e < Ee) ? (htype)emb[(size_t)tok * Ee + e] : (htype)0.f;
}

// ---------------- Viterbi (forward + backtrack), one block per batch row ----------------
__global__ void viterbi_k(const float* __restrict__ emis, const int* __restrict__ seq_len,
                          const float* __restrict__ trans, unsigned char* __restrict__ btrk,
                          float* __restrict__ pred) {
  int b = blockIdx.x, j = threadIdx.x;
  __shared__ float sc[Cpad];
  __shared__ float tr[Cc * Cc];
  for (int k = j; k < Cc * Cc; k += 32) tr[k] = trans[k];
  if (j < Cc) sc[j] = 0.f;
  __syncthreads();
  int L = seq_len[b];
  for (int t = 0; t < Tt; ++t) {
    float best = 0.f; int bi = 0;
    if (j < Cc) {
      best = sc[0] + tr[j]; bi = 0;
      for (int i = 1; i < Cc; ++i) {
        float v = sc[i] + tr[i * Cc + j];
        if (v > best) { best = v; bi = i; }
      }
    }
    __syncthreads();
    if (j < Cc) {
      btrk[(size_t)(b * Tt + t) * Cpad + j] = (unsigned char)bi;
      if (t < L) sc[j] = best + emis[(size_t)(b * Tt + t) * Cpad + j];
    }
    __syncthreads();
  }
  if (j < Cc) sc[j] += tr[j * Cc + STOP_TAG];
  __syncthreads();
  if (j == 0) {
    int cur = 0; float best = sc[0];
    for (int i = 1; i < Cc; ++i) if (sc[i] > best) { best = sc[i]; cur = i; }
    for (int l = Tt - 1; l >= 0; --l) {
      pred[b * Tt + l] = (l < L) ? (float)cur : 0.f;
      if (l >= 1 && l <= L - 1) cur = btrk[(size_t)(b * Tt + l) * Cpad + cur];
    }
  }
}

// ---------------- CRF NLL: forward logsumexp + gold score, per batch row ----------------
__global__ void nll_fwd(const float* __restrict__ emis, const int* __restrict__ seq_len,
                        const int* __restrict__ tags, const float* __restrict__ trans,
                        float* __restrict__ perb) {
  int b = blockIdx.x, j = threadIdx.x;
  __shared__ float sc[Cpad];
  __shared__ float tr[Cc * Cc];
  for (int k = j; k < Cc * Cc; k += 32) tr[k] = trans[k];
  if (j < Cc) sc[j] = NEGVAL;
  __syncthreads();
  int L = seq_len[b];
  for (int t = 0; t < Tt; ++t) {
    float nv = 0.f;
    if (j < Cc) {
      float m = -3.4e38f;
      for (int i = 0; i < Cc; ++i) { float v = sc[i] + tr[i * Cc + j]; m = v > m ? v : m; }
      float s = 0.f;
      for (int i = 0; i < Cc; ++i) s += expf(sc[i] + tr[i * Cc + j] - m);
      nv = emis[(size_t)(b * Tt + t) * Cpad + j] + m + logf(s);
    }
    __syncthreads();
    if (j < Cc && t < L) sc[j] = nv;
    __syncthreads();
  }
  if (j == 0) {
    float m = -3.4e38f;
    for (int i = 0; i < Cc; ++i) { float v = sc[i] + tr[i * Cc + STOP_TAG]; if (v > m) m = v; }
    float s = 0.f;
    for (int i = 0; i < Cc; ++i) s += expf(sc[i] + tr[i * Cc + STOP_TAG] - m);
    float fwd = m + logf(s);
    float gold = 0.f; int prev = START_TAG;
    for (int t = 0; t < L; ++t) {
      int tg = tags[b * Tt + t];
      gold += emis[(size_t)(b * Tt + t) * Cpad + tg] + tr[prev * Cc + tg];
      prev = tg;
    }
    gold += tr[tags[b * Tt + (L - 1)] * Cc + STOP_TAG];
    perb[b] = fwd - gold;
  }
}
__global__ void nll_red(const float* perb, float* out) {
  if (blockIdx.x == 0 && threadIdx.x == 0) {
    float s = 0.f;
    for (int b = 0; b < Bb; ++b) s += perb[b];
    out[0] = s / (float)Bb;
  }
}

// ---------------- host orchestration ----------------
extern "C" void kernel_launch(void* const* d_in, const int* in_sizes, int n_in,
                              void* d_out, int out_size, void* d_ws, size_t ws_size,
                              hipStream_t stream) {
  (void)in_sizes; (void)n_in; (void)out_size; (void)ws_size;
  const int*   tokens  = (const int*)d_in[0];
  const int*   seq_len = (const int*)d_in[1];
  const int*   tags    = (const int*)d_in[3];
  const float* emb     = (const float*)d_in[4];
  const float* wih[4]  = { (const float*)d_in[5],  (const float*)d_in[9],
                           (const float*)d_in[13], (const float*)d_in[17] };
  const float* whh[4]  = { (const float*)d_in[6],  (const float*)d_in[10],
                           (const float*)d_in[14], (const float*)d_in[18] };
  const float* bih[4]  = { (const float*)d_in[7],  (const float*)d_in[11],
                           (const float*)d_in[15], (const float*)d_in[19] };
  const float* bhh[4]  = { (const float*)d_in[8],  (const float*)d_in[12],
                           (const float*)d_in[16], (const float*)d_in[20] };
  const float* fc_w    = (const float*)d_in[21];
  const float* fc_b    = (const float*)d_in[22];
  const float* trans   = (const float*)d_in[23];

  char* p = (char*)d_ws;
  auto alloc = [&](size_t bytes) { void* r = (void*)p; p += (bytes + 255) & ~(size_t)255; return r; };
  htype* XA   = (htype*)alloc((size_t)BT * XSTR * 2);
  htype* XB   = (htype*)alloc((size_t)BT * XSTR * 2);
  htype* GX   = (htype*)alloc((size_t)BT * Gg * 2);
  htype* WIH16[4]; htype* WHH16[4]; float* BSUM[4];
  for (int i = 0; i < 4; ++i) {
    int kw = (i < 2) ? Ee : 1024;
    WIH16[i] = (htype*)alloc((size_t)Gg * kw * 2);
    WHH16[i] = (htype*)alloc((size_t)Gg * Hh * 2);
    BSUM[i]  = (float*)alloc((size_t)Gg * 4);
  }
  htype* FCW16 = (htype*)alloc((size_t)Cpad * 1024 * 2);
  float* FCB16 = (float*)alloc(Cpad * 4);
  float* Hst   = (float*)alloc((size_t)Bb * Hh * 4);
  float* Cst   = (float*)alloc((size_t)Bb * Hh * 4);
  htype* H16   = (htype*)alloc((size_t)Bb * Hh * 2);
  float* GATES = (float*)alloc((size_t)Bb * Gg * 4);
  float* EMIS  = (float*)alloc((size_t)BT * Cpad * 4);
  unsigned char* BTRK = (unsigned char*)alloc((size_t)BT * Cpad);
  float* PERB  = (float*)alloc(Bb * 4);

  // ---- weight conversion to f16 + bias sums ----
  for (int i = 0; i < 4; ++i) {
    int kw = (i < 2) ? Ee : 1024;
    conv_f16<<<1024, 256, 0, stream>>>(wih[i], WIH16[i], Gg * kw);
    conv_f16<<<1024, 256, 0, stream>>>(whh[i], WHH16[i], Gg * Hh);
    bias_sum<<<8, 256, 0, stream>>>(bih[i], bhh[i], BSUM[i]);
  }
  pad_fcw<<<64, 256, 0, stream>>>(fc_w, FCW16);
  pad_fcb<<<1, 32, 0, stream>>>(fc_b, FCB16);

  // ---- embedding gather (zero-padded to Epad) ----
  gather_emb<<<BT, Epad, 0, stream>>>(tokens, emb, XA);

  // ---- two BiLSTM layers, one direction at a time (GX buffer reused) ----
  for (int l = 0; l < 2; ++l) {
    const htype* inX = l ? XB : XA;
    htype* outX = l ? XA : XB;
    int Kw = l ? 1024 : Ee;
    for (int d = 0; d < 2; ++d) {
      int idx = l * 2 + d;
      if (l == 0)
        gemm_gx<320, 320><<<dim3(BT / 32, Gg / 64), 256, 0, stream>>>(
            inX, WIH16[idx], BSUM[idx], GX, Kw);
      else
        gemm_gx<256, 1024><<<dim3(BT / 32, Gg / 64), 256, 0, stream>>>(
            inX, WIH16[idx], BSUM[idx], GX, Kw);
      (void)hipMemsetAsync(Hst, 0, (size_t)Bb * Hh * 4, stream);
      (void)hipMemsetAsync(Cst, 0, (size_t)Bb * Hh * 4, stream);
      (void)hipMemsetAsync(H16, 0, (size_t)Bb * Hh * 2, stream);
      for (int t = 0; t < Tt; ++t) {
        int t_eff = d ? (Tt - 1 - t) : t;
        gemm_rec<<<dim3(Bb / 16, Gg / 128), 256, 0, stream>>>(H16, WHH16[idx], GX,
                                                              GATES, t_eff);
        lstm_cell<<<(Bb * Hh) / 256, 256, 0, stream>>>(GATES, Hst, Cst, H16, outX,
                                                       seq_len, t_eff, d);
      }
    }
  }

  // ---- emissions, Viterbi decode, CRF NLL ----
  gemm_fc<<<BT / 128, 256, 0, stream>>>(XA, FCW16, FCB16, EMIS);
  viterbi_k<<<Bb, 32, 0, stream>>>(EMIS, seq_len, trans, BTRK, (float*)d_out);
  nll_fwd<<<Bb, 32, 0, stream>>>(EMIS, seq_len, tags, trans, PERB);
  nll_red<<<1, 32, 0, stream>>>(PERB, (float*)d_out + BT);
}
